// SplineCouplingLayer_88046829568178
// MI455X (gfx1250) — matrix operand order repaired and checked by reading, not compiled, not run on hardware
//
#include <hip/hip_runtime.h>
#include <hip/hip_bf16.h>

typedef _Float16 v16h __attribute__((ext_vector_type(16)));
typedef float    v8f  __attribute__((ext_vector_type(8)));

// explicit global (addrspace 1) pointers so weight streams lower to
// global_load_b128 (LOADcnt only) instead of flat_load_b128 (LOADcnt+DScnt)
typedef const _Float16 __attribute__((address_space(1)))* gf16cp;
typedef _Float16       __attribute__((address_space(1)))* gf16p;
typedef const float    __attribute__((address_space(1)))* gf32cp;
typedef float          __attribute__((address_space(1)))* gf32p;
typedef const v16h     __attribute__((address_space(1)))* gv16cp;

#define BATCH        131072
#define NUM_FEAT     10
#define NUM_ID       5
#define NUM_T        5
#define NUM_BINS     8
#define EMBED        256
#define LINEAR       512
#define IN_DIM       261      // 5 + 256
#define K1PAD        288      // IN_DIM padded to multiple of 32
#define N2PAD        48       // max(40,35) padded to multiple of 16
#define TB           2.0f
#define MIN_BW       0.001f
#define MIN_D        0.001f
#define INV_SCALE    0.044194173824159216f   // 1/sqrt(512)
#define MTILE        32

// ---------------- ws layout (f16 transposed weights) ----------------
#define W1T_ELEMS   (LINEAR * K1PAD)          // 147456
#define W2T_ELEMS   (N2PAD * LINEAR)          // 24576
#define W2T_BYTE_OFF (3 * W1T_ELEMS * 2)      // 884736

__global__ __launch_bounds__(256)
void prep_weights(const float* __restrict__ w1w, const float* __restrict__ w1h,
                  const float* __restrict__ w1d, const float* __restrict__ w2w,
                  const float* __restrict__ w2h, const float* __restrict__ w2d,
                  _Float16* __restrict__ w1t_, _Float16* __restrict__ w2t_) {
    gf32cp W1[3] = {(gf32cp)w1w, (gf32cp)w1h, (gf32cp)w1d};
    gf32cp W2[3] = {(gf32cp)w2w, (gf32cp)w2h, (gf32cp)w2d};
    gf16p  w1t = (gf16p)w1t_;
    gf16p  w2t = (gf16p)w2t_;
    const int NOUT[3] = {40, 40, 35};

    int total1 = 3 * W1T_ELEMS;
    for (int i = blockIdx.x * blockDim.x + threadIdx.x; i < total1;
         i += gridDim.x * blockDim.x) {
        int h = i / W1T_ELEMS;
        int r = i - h * W1T_ELEMS;
        int n = r / K1PAD;
        int k = r - n * K1PAD;
        float v = (k < IN_DIM) ? W1[h][k * LINEAR + n] : 0.0f;   // w1 is (261,512)
        w1t[i] = (_Float16)v;
    }
    int total2 = 3 * W2T_ELEMS;
    for (int i = blockIdx.x * blockDim.x + threadIdx.x; i < total2;
         i += gridDim.x * blockDim.x) {
        int h = i / W2T_ELEMS;
        int r = i - h * W2T_ELEMS;
        int n = r / LINEAR;
        int k = r - n * LINEAR;
        int no = NOUT[h];
        float v = (n < no) ? W2[h][k * no + n] : 0.0f;           // w2 is (512,NOUT)
        w2t[i] = (_Float16)v;
    }
}

__device__ __forceinline__ float softplusf(float x) {
    return (x > 20.0f) ? x : log1pf(__expf(x));
}

// ---------------- fused MLP(x3) + spline kernel ----------------
// LDS map (static 60 KB):
//   [0      .. 18432)  Xs   : 32 x 288 f16   (reused for Pd + ladS later)
//   [18432  .. 51200)  Hs   : 32 x 512 f16
//   [51200  .. 56320)  Pw   : 32 x 40  f32
//   [56320  .. 61440)  Ph   : 32 x 40  f32
//   Pd   = smem + 0     : 32 x 35 f32 (4480 B, inside dead Xs region)
//   ladS = smem + 4480  : 32 x 5  f32 (640 B)
__global__ __launch_bounds__(256)
void spline_fused(const float* __restrict__ inp_,  const float* __restrict__ emb_,
                  const float* __restrict__ b1w,  const float* __restrict__ b2w,
                  const float* __restrict__ b1h,  const float* __restrict__ b2h,
                  const float* __restrict__ b1d,  const float* __restrict__ b2d,
                  const _Float16* __restrict__ w1t, const _Float16* __restrict__ w2t,
                  float* __restrict__ out_) {
    __shared__ __align__(32) unsigned char smem[61440];
    _Float16* Xs = (_Float16*)smem;
    _Float16* Hs = (_Float16*)(smem + 18432);
    float*    Pw = (float*)(smem + 51200);
    float*    Ph = (float*)(smem + 56320);
    float*    Pd = (float*)smem;            // reuse (head 2 only)
    float*    ladS = (float*)(smem + 4480); // reuse (spline phase only)

    gf32cp inp = (gf32cp)inp_;
    gf32cp emb = (gf32cp)emb_;
    gf32p  out = (gf32p)out_;

    const int tid  = threadIdx.x;
    const int lane = tid & 31;
    const int wv   = tid >> 5;
    const int half = lane >> 4;
    const int l16  = lane & 15;
    const long rowBase = (long)blockIdx.x * MTILE;

    // ---- stage X tile: concat(inputs[:, :5], embedded) -> f16, K padded ----
    for (int i = tid; i < MTILE * K1PAD; i += 256) {
        int r = i / K1PAD;
        int k = i - r * K1PAD;
        long gr = rowBase + r;
        float v = 0.0f;
        if (k < NUM_ID)       v = inp[gr * NUM_FEAT + k];
        else if (k < IN_DIM)  v = emb[gr * EMBED + (k - NUM_ID)];
        Xs[i] = (_Float16)v;
    }
    __syncthreads();

    const gf16cp W1T[3] = {(gf16cp)w1t, (gf16cp)w1t + W1T_ELEMS,
                           (gf16cp)w1t + 2 * W1T_ELEMS};
    const gf16cp W2T[3] = {(gf16cp)w2t, (gf16cp)w2t + W2T_ELEMS,
                           (gf16cp)w2t + 2 * W2T_ELEMS};
    const gf32cp B1[3] = {(gf32cp)b1w, (gf32cp)b1h, (gf32cp)b1d};
    const gf32cp B2[3] = {(gf32cp)b2w, (gf32cp)b2h, (gf32cp)b2d};
    float* P[3]   = {Pw, Ph, Pd};
    const int   NOUT[3] = {40, 40, 35};
    const float MUL[3]  = {INV_SCALE, INV_SCALE, 1.0f};

    #pragma unroll
    for (int hd = 0; hd < 3; ++hd) {
        // ================= GEMM1: H = elu(X @ W1 + b1) =================
        // 8 waves: wm = wv&1 (M subtile of 16), wc = wv>>1 (N column of 16)
        const int m0 = (wv & 1) * 16;
        const int wc = wv >> 1;

        // hoist all 9 A-fragments (rows m0..m0+15, K 0..287) into registers
        v16h a[9];
        #pragma unroll
        for (int kc = 0; kc < 9; ++kc) {
            const _Float16* base = &Xs[(m0 + l16) * K1PAD + kc * 32];
            #pragma unroll
            for (int g = 0; g < 8; ++g) {
                // A layout matches B: K = half*16+2g(+p) (contiguous)
                int kp = half * 16 + 2 * g;
                a[kc][2 * g]     = base[kp];
                a[kc][2 * g + 1] = base[kp + 1];
            }
        }

        // hoist per-column biases out of the N loop
        float biasArr[8];
        #pragma unroll
        for (int np = 0; np < 8; ++np)
            biasArr[np] = B1[hd][np * 64 + wc * 16 + l16];

        const gf16cp wT = W1T[hd];
        #pragma unroll 2
        for (int np = 0; np < 8; ++np) {
            const int n0 = np * 64 + wc * 16;
            v8f acc = {};
            #pragma unroll
            for (int kc = 0; kc < 9; ++kc) {
                // B layout: lane -> N = n0+l16; K = kc*32 + half*16 + e (contiguous 32B)
                gf16cp bp = wT + (n0 + l16) * K1PAD + kc * 32 + half * 16;
                v16h b = *(gv16cp)bp;
                acc = __builtin_amdgcn_wmma_f32_16x16x32_f16(
                        false, a[kc], false, b, (short)0, acc, false, false);
            }
            const int n = n0 + l16;
            const float bias = biasArr[np];
            #pragma unroll
            for (int v = 0; v < 8; ++v) {
                float x = acc[v] + bias;
                float e = (x > 0.0f) ? x : (__expf(x) - 1.0f);   // elu
                Hs[(m0 + v + 8 * half) * LINEAR + n] = (_Float16)e;
            }
        }
        __syncthreads();

        // ================= GEMM2: P = (H @ W2 + b2) * mul =================
        // 6 fragment tiles (2 M x 3 N) on waves 0..5
        if (wv < 6) {
            const int m0b = (wv & 1) * 16;
            const int n0  = (wv >> 1) * 16;
            const gf16cp w2p = W2T[hd];
            v8f acc = {};
            #pragma unroll 4
            for (int kc = 0; kc < 16; ++kc) {
                v16h a2;
                const _Float16* base = &Hs[(m0b + l16) * LINEAR + kc * 32];
                #pragma unroll
                for (int g = 0; g < 8; ++g) {
                    int kp = half * 16 + 2 * g;
                    a2[2 * g]     = base[kp];
                    a2[2 * g + 1] = base[kp + 1];
                }
                gf16cp bp = w2p + (n0 + l16) * LINEAR + kc * 32 + half * 16;
                v16h b = *(gv16cp)bp;
                acc = __builtin_amdgcn_wmma_f32_16x16x32_f16(
                        false, a2, false, b, (short)0, acc, false, false);
            }
            const int n = n0 + l16;
            const int no = NOUT[hd];
            if (n < no) {
                const float bias = B2[hd][n];
                const float m = MUL[hd];
                float* dst = P[hd];
                #pragma unroll
                for (int v = 0; v < 8; ++v)
                    dst[(m0b + v + 8 * half) * no + n] = (acc[v] + bias) * m;
            }
        }
        __syncthreads();
    }

    // ================= spline: one thread per (row, t) =================
    if (tid < MTILE * NUM_T) {
        const int r = tid / NUM_T;
        const int t = tid - r * NUM_T;
        const long gr = rowBase + r;
        const float xin = inp[gr * NUM_FEAT + NUM_ID + t];
        const bool inside = (xin >= -TB) && (xin <= TB);
        const float xs = fminf(fmaxf(xin, -TB), TB);

        float uw[NUM_BINS], uh[NUM_BINS];
        #pragma unroll
        for (int k = 0; k < NUM_BINS; ++k) {
            uw[k] = Pw[r * 40 + t * NUM_BINS + k];
            uh[k] = Ph[r * 40 + t * NUM_BINS + k];
        }
        float cw[NUM_BINS + 1], chh[NUM_BINS + 1], wid[NUM_BINS], hei[NUM_BINS];
        {
            float mx = uw[0];
            #pragma unroll
            for (int k = 1; k < NUM_BINS; ++k) mx = fmaxf(mx, uw[k]);
            float e[NUM_BINS], s = 0.0f;
            #pragma unroll
            for (int k = 0; k < NUM_BINS; ++k) { e[k] = __expf(uw[k] - mx); s += e[k]; }
            float inv = 1.0f / s, accum = 0.0f;
            cw[0] = -TB;
            #pragma unroll
            for (int k = 0; k < NUM_BINS; ++k) {
                accum += MIN_BW + (1.0f - MIN_BW * NUM_BINS) * e[k] * inv;
                cw[k + 1] = 2.0f * TB * accum - TB;
            }
            cw[NUM_BINS] = TB;
            #pragma unroll
            for (int k = 0; k < NUM_BINS; ++k) wid[k] = cw[k + 1] - cw[k];
        }
        {
            float mx = uh[0];
            #pragma unroll
            for (int k = 1; k < NUM_BINS; ++k) mx = fmaxf(mx, uh[k]);
            float e[NUM_BINS], s = 0.0f;
            #pragma unroll
            for (int k = 0; k < NUM_BINS; ++k) { e[k] = __expf(uh[k] - mx); s += e[k]; }
            float inv = 1.0f / s, accum = 0.0f;
            chh[0] = -TB;
            #pragma unroll
            for (int k = 0; k < NUM_BINS; ++k) {
                accum += MIN_BW + (1.0f - MIN_BW * NUM_BINS) * e[k] * inv;
                chh[k + 1] = 2.0f * TB * accum - TB;
            }
            chh[NUM_BINS] = TB;
            #pragma unroll
            for (int k = 0; k < NUM_BINS; ++k) hei[k] = chh[k + 1] - chh[k];
        }
        float dv[NUM_BINS + 1];
        dv[0] = 1.0f; dv[NUM_BINS] = 1.0f;   // boundary pads reduce to exactly 1.0
        #pragma unroll
        for (int j = 0; j < NUM_BINS - 1; ++j)
            dv[j + 1] = MIN_D + softplusf(Pd[r * 35 + t * (NUM_BINS - 1) + j]);

        int bin = 0;
        #pragma unroll
        for (int k = 1; k <= NUM_BINS; ++k) bin += (xs >= cw[k]) ? 1 : 0;
        bin = (bin > NUM_BINS - 1) ? (NUM_BINS - 1) : bin;

        const float icw = cw[bin], iw = wid[bin];
        const float ich = chh[bin], ih = hei[bin];
        const float delta = ih / iw;
        const float d0 = dv[bin], d1 = dv[bin + 1];
        const float theta = (xs - icw) / iw;
        const float omt   = 1.0f - theta;
        const float t1m   = theta * omt;
        const float numer = ih * (delta * theta * theta + d0 * t1m);
        const float denom = delta + (d0 + d1 - 2.0f * delta) * t1m;
        float y = ich + numer / denom;
        const float dnum = delta * delta *
                           (d1 * theta * theta + 2.0f * delta * t1m + d0 * omt * omt);
        float lad = __logf(dnum) - 2.0f * __logf(denom);
        y   = inside ? y   : xin;
        lad = inside ? lad : 0.0f;

        out[gr * NUM_FEAT + NUM_ID + t] = y;
        ladS[r * NUM_T + t] = lad;
    }
    // identity passthrough
    for (int i = tid; i < MTILE * NUM_ID; i += 256) {
        int r = i / NUM_ID, c = i - r * NUM_ID;
        long gr = rowBase + r;
        out[gr * NUM_FEAT + c] = inp[gr * NUM_FEAT + c];
    }
    __syncthreads();
    // lad sum over t
    if (tid < MTILE) {
        long gr = rowBase + tid;
        float s = 0.0f;
        #pragma unroll
        for (int t = 0; t < NUM_T; ++t) s += ladS[tid * NUM_T + t];
        out[(long)BATCH * NUM_FEAT + gr] = s;
    }
}

extern "C" void kernel_launch(void* const* d_in, const int* in_sizes, int n_in,
                              void* d_out, int out_size, void* d_ws, size_t ws_size,
                              hipStream_t stream) {
    const float* inp  = (const float*)d_in[0];
    const float* emb  = (const float*)d_in[1];
    const float* w1w  = (const float*)d_in[2];
    const float* b1w  = (const float*)d_in[3];
    const float* w2w  = (const float*)d_in[4];
    const float* b2w  = (const float*)d_in[5];
    const float* w1h  = (const float*)d_in[6];
    const float* b1h  = (const float*)d_in[7];
    const float* w2h  = (const float*)d_in[8];
    const float* b2h  = (const float*)d_in[9];
    const float* w1d  = (const float*)d_in[10];
    const float* b1d  = (const float*)d_in[11];
    const float* w2d  = (const float*)d_in[12];
    const float* b2d  = (const float*)d_in[13];
    float* out = (float*)d_out;

    _Float16* w1t = (_Float16*)d_ws;                                   // 3 x 512 x 288
    _Float16* w2t = (_Float16*)((unsigned char*)d_ws + W2T_BYTE_OFF);  // 3 x 48 x 512

    prep_weights<<<512, 256, 0, stream>>>(w1w, w1h, w1d, w2w, w2h, w2d, w1t, w2t);
    spline_fused<<<BATCH / MTILE, 256, 0, stream>>>(
        inp, emb, b1w, b2w, b1h, b2h, b1d, b2d, w1t, w2t, out);
}
